// RGCNModel_44212393345114
// MI455X (gfx1250) — compile-verified, matrix operand order checked by reading, not run on hardware
//
#include <hip/hip_runtime.h>

// ---------------------------------------------------------------------------
// RGCN (3-layer, mean-agg) + BN + ReLU + pool + MLP for MI455X (gfx1250).
// GEMMs: bf16 v_wmma_f32_16x16x32_bf16, f32 accum, 128x128x32 tiles.
// A tiles are DMA'd into LDS by the Tensor Data Mover (tensor_load_to_lds,
// TENSORcnt) when the builtin is available; B tiles are staged transposed so
// fragments load as two contiguous ds_load_b128 per ISA operand layout.
// Relation conv is transform-first: Hr = h @ W_r (bf16), then per-edge
// scatter Hr[src] * (1/cnt[r,dst]) into conv[dst] with f32 atomics; the
// 102 MB accumulator fits in the 192 MB L2 so atomics stay L2-resident.
// ---------------------------------------------------------------------------

typedef __attribute__((ext_vector_type(16))) __bf16 v16bf;
typedef __attribute__((ext_vector_type(8)))  float  v8f;
typedef unsigned int u32x4 __attribute__((ext_vector_type(4)));
typedef int          i32x4 __attribute__((ext_vector_type(4)));
typedef int          i32x8 __attribute__((ext_vector_type(8)));

#if __has_builtin(__builtin_amdgcn_tensor_load_to_lds) && \
    __has_builtin(__builtin_amdgcn_s_wait_tensorcnt)
#define USE_TDM 1
#else
#define USE_TDM 0
#endif

#define DEV __device__ __forceinline__

constexpr int   NN   = 100000;  // nodes
constexpr int   EE   = 800000;  // edges
constexpr int   RR   = 8;       // relations
constexpr int   DINC = 128;     // input features
constexpr int   DDC  = 256;     // hidden dim
constexpr int   GGC  = 64;      // graphs
constexpr float EPSV = 1e-5f;

DEV unsigned short f2bf(float f) {
  unsigned int u = __float_as_uint(f);
  u += 0x7FFFu + ((u >> 16) & 1u);           // round-to-nearest-even
  return (unsigned short)(u >> 16);
}
DEV float bf2f(unsigned short h) {
  return __uint_as_float(((unsigned int)h) << 16);
}

// ------------------------------- elementwise -------------------------------

__global__ void k_zero_f32(float* p, long n) {
  long i = (long)blockIdx.x * blockDim.x + threadIdx.x;
  if (i < n) p[i] = 0.0f;
}

__global__ void k_cvt_bf16(const float* __restrict__ s,
                           unsigned short* __restrict__ d, long n) {
  long i = (long)blockIdx.x * blockDim.x + threadIdx.x;
  if (i < n) d[i] = f2bf(s[i]);
}

__global__ void k_add_res(float* __restrict__ conv,
                          const float* __restrict__ res, long n) {
  long i = (long)blockIdx.x * blockDim.x + threadIdx.x;
  if (i < n) conv[i] += res[i];
}

// ------------------------------- edge counts -------------------------------

__global__ void k_count_edges(const int* __restrict__ dst,
                              const int* __restrict__ et,
                              float* __restrict__ cnt) {
  int e = blockIdx.x * blockDim.x + threadIdx.x;
  if (e < EE) atomicAdd(&cnt[et[e] * NN + dst[e]], 1.0f);
}

__global__ void k_inv_cnt(float* __restrict__ cnt, long n) {
  long i = (long)blockIdx.x * blockDim.x + threadIdx.x;
  if (i < n) cnt[i] = 1.0f / fmaxf(cnt[i], 1.0f);
}

// ------------------------- per-relation edge scatter -----------------------
// block = one edge, 256 threads = 256 features. Early-exit if wrong relation.
__global__ __launch_bounds__(256)
void k_scatter_rel(const unsigned short* __restrict__ Hr,
                   const int* __restrict__ src, const int* __restrict__ dst,
                   const int* __restrict__ et, const float* __restrict__ inv,
                   float* __restrict__ conv, int r) {
  int e = blockIdx.x;
  if (et[e] != r) return;
  int s = src[e], d = dst[e];
  float w = inv[r * NN + d];
  int t = threadIdx.x;
  float v = bf2f(Hr[(size_t)s * DDC + t]) * w;
  atomicAdd(&conv[(size_t)d * DDC + t], v);
}

// -------------------------------- BatchNorm --------------------------------

__global__ __launch_bounds__(256)
void k_bn_stats(const float* __restrict__ h, float* __restrict__ sum,
                float* __restrict__ sumsq) {
  int c = threadIdx.x;                 // column 0..255 (coalesced row reads)
  float s = 0.0f, ss = 0.0f;
  for (int row = blockIdx.x; row < NN; row += gridDim.x) {
    float v = h[(size_t)row * DDC + c];
    s += v; ss += v * v;
  }
  atomicAdd(&sum[c], s);
  atomicAdd(&sumsq[c], ss);
}

__global__ void k_bn_apply_relu(const float* __restrict__ conv,
                                const float* __restrict__ sum,
                                const float* __restrict__ sumsq,
                                const float* __restrict__ gamma,
                                const float* __restrict__ beta,
                                float* __restrict__ outf,
                                unsigned short* __restrict__ outbf) {
  long i = (long)blockIdx.x * blockDim.x + threadIdx.x;
  if (i >= (long)NN * DDC) return;
  int c = (int)(i & (DDC - 1));
  const float invN = 1.0f / (float)NN;
  float m   = sum[c] * invN;
  float var = sumsq[c] * invN - m * m;
  float rs  = rsqrtf(var + EPSV);
  float v   = (conv[i] - m) * rs * gamma[c] + beta[c];
  v = fmaxf(v, 0.0f);
  outf[i] = v;
  outbf[i] = f2bf(v);
}

// ---------------------------------- pooling --------------------------------

__global__ void k_pool_accum(const float* __restrict__ x3,
                             const int* __restrict__ batch,
                             float* __restrict__ psum) {
  long i = (long)blockIdx.x * blockDim.x + threadIdx.x;
  if (i >= (long)NN * DDC) return;
  int row = (int)(i >> 8);
  int c   = (int)(i & 255);
  atomicAdd(&psum[batch[row] * DDC + c], x3[i]);
}

__global__ void k_pool_cnt(const int* __restrict__ batch,
                           float* __restrict__ pcnt) {
  int n = blockIdx.x * blockDim.x + threadIdx.x;
  if (n < NN) atomicAdd(&pcnt[batch[n]], 1.0f);
}

__global__ void k_pool_div(const float* __restrict__ psum,
                           const float* __restrict__ pcnt,
                           unsigned short* __restrict__ pooledbf) {
  int i = blockIdx.x * blockDim.x + threadIdx.x;
  if (i >= GGC * DDC) return;
  int g = i >> 8;
  float v = psum[i] / fmaxf(pcnt[g], 1.0f);
  pooledbf[i] = f2bf(v);
}

// ---------------------- final 256 -> 2 classifier layer --------------------

__global__ void k_final_fc(const float* __restrict__ hidden,
                           const float* __restrict__ W,
                           const float* __restrict__ b,
                           float* __restrict__ out) {
  int i = blockIdx.x * blockDim.x + threadIdx.x;
  if (i >= GGC * 2) return;
  int g = i >> 1, c = i & 1;
  float s = b[c];
  for (int k = 0; k < DDC; ++k) s += hidden[g * DDC + k] * W[k * 2 + c];
  out[i] = s;
}

// ------------------------------ WMMA bf16 GEMM -----------------------------
// C[M,Nn] (f32 and/or bf16) = A[M,K]_bf16 @ B[K,Nn]_bf16 (+bias) (+relu)
// Tile 128x128, BK=32. 8 waves; each wave (4x2) computes 2x4 16x16 WMMA
// tiles. A tile is DMA'd to LDS via the Tensor Data Mover by wave 0 (OOB
// rows beyond M auto-zero via tensor_dim1); B is transposed into LDS so both
// fragment types load as two contiguous ds b128 per the ISA bf16 layout
// (lanes 0-15: K 0..7 / 16..23, lanes 16-31: K 8..15 / 24..31).
// Requires K % 32 == 0 and Nn % 128 == 0 (true for all uses here).
__global__ __launch_bounds__(256)
void k_gemm_bf16(const unsigned short* __restrict__ A,
                 const unsigned short* __restrict__ B,
                 const float* __restrict__ bias,
                 float* __restrict__ Cf,
                 unsigned short* __restrict__ Cbf,
                 int M, int Nn, int K, int relu) {
  __shared__ unsigned short As[128][32];
  __shared__ unsigned short Bs[128][32];   // transposed: [col][k]

  const int tid  = threadIdx.x;
  const int lane = tid & 31;
  const int wave = tid >> 5;
  const int wm   = wave & 3;     // 0..3  -> 32-row groups
  const int wn   = wave >> 2;    // 0..1  -> 64-col groups
  const int lr   = lane & 15;
  const int lh   = lane >> 4;    // K-half / M-half selector
  const int blockRow = blockIdx.x * 128;
  const int colBase  = blockIdx.y * 128;

  v8f acc[2][4];
#pragma unroll
  for (int i = 0; i < 2; ++i)
#pragma unroll
    for (int j = 0; j < 4; ++j)
#pragma unroll
      for (int v = 0; v < 8; ++v) acc[i][j][v] = 0.0f;

#if USE_TDM
  // Uniform descriptor pieces for the A-tile TDM transfer.
  const unsigned ldsA = (unsigned)(size_t)(&As[0][0]);  // LDS byte offset
  int rows = M - blockRow;
  if (rows > 128) rows = 128;
#endif

  for (int kt = 0; kt < K; kt += 32) {
#if USE_TDM
    // ---- A tile via Tensor Data Mover: 2D tile, 32 bf16 x 128 rows ----
    if (wave == 0) {
      unsigned long long ga =
          (unsigned long long)(size_t)(A + (size_t)blockRow * K + kt);
      u32x4 g0;
      g0[0] = 1u;                                    // count=1 (user D#)
      g0[1] = ldsA;                                  // lds_addr (bytes)
      g0[2] = (unsigned)(ga & 0xffffffffu);          // global_addr[31:0]
      g0[3] = (unsigned)((ga >> 32) & 0x01ffffffu)   // global_addr[56:32]
              | (2u << 30);                          // type = 2 ("image")
      i32x8 g1;
      g1[0] = 1 << 16;                               // data_size=1 -> 2 bytes
      g1[1] = (int)(32u << 16);                      // tensor_dim0 = 32 (lo16)
      g1[2] = (int)(((unsigned)rows & 0xffffu) << 16); // dim0 hi=0 | dim1 lo
      g1[3] = (int)((((unsigned)rows >> 16) & 0xffffu) // dim1 hi
                    | (32u << 16));                  // tile_dim0 = 32
      g1[4] = 128;                                   // tile_dim1=128, dim2=0
      g1[5] = (int)(unsigned)K;                      // dim0_stride[31:0] = K
      g1[6] = 0;                                     // stride hi / dim1_stride
      g1[7] = 0;
      i32x4 g2 = {0, 0, 0, 0};
      i32x4 g3 = {0, 0, 0, 0};
#if defined(__clang_major__) && __clang_major__ >= 23
      i32x8 g4 = {0, 0, 0, 0, 0, 0, 0, 0};
      __builtin_amdgcn_tensor_load_to_lds(g0, g1, g2, g3, g4, 0);
#else
      __builtin_amdgcn_tensor_load_to_lds(g0, g1, g2, g3, 0);
#endif
      __builtin_amdgcn_s_wait_tensorcnt(0);
    }
#else
    // ---- A tile: 128x32 bf16 = 512 uint4; 2 per thread, guarded on M ----
#pragma unroll
    for (int i = 0; i < 2; ++i) {
      int u    = tid + i * 256;
      int row  = u >> 2;
      int c8   = (u & 3) << 3;
      int grow = blockRow + row;
      uint4 val = {0u, 0u, 0u, 0u};
      if (grow < M) {
        const unsigned short* gp = A + (size_t)grow * K + kt + c8;
        val = *(const uint4*)gp;
      }
      *(uint4*)(&As[row][c8]) = val;
    }
#endif
    // ---- B tile: read row-major [k][col], store transposed [col][k] ----
    {
      int k  = tid & 31;
      int cg = tid >> 5;
      const unsigned short* bp = B + (size_t)(kt + k) * Nn + colBase + cg * 16;
      if (kt + 32 < K) __builtin_prefetch(bp + 32 * Nn, 0, 0);  // next k-tile
#pragma unroll
      for (int e = 0; e < 16; ++e) Bs[cg * 16 + e][k] = bp[e];
    }
    __syncthreads();

    union Frag { v16bf v; uint4 q[2]; };
    Frag a[2], b[4];
#pragma unroll
    for (int am = 0; am < 2; ++am) {
      const unsigned short* p = &As[wm * 32 + am * 16 + lr][lh * 8];
      a[am].q[0] = *(const uint4*)p;
      a[am].q[1] = *(const uint4*)(p + 16);
    }
#pragma unroll
    for (int bn = 0; bn < 4; ++bn) {
      const unsigned short* p = &Bs[wn * 64 + bn * 16 + lr][lh * 8];
      b[bn].q[0] = *(const uint4*)p;
      b[bn].q[1] = *(const uint4*)(p + 16);
    }
#pragma unroll
    for (int am = 0; am < 2; ++am)
#pragma unroll
      for (int bn = 0; bn < 4; ++bn)
        acc[am][bn] = __builtin_amdgcn_wmma_f32_16x16x32_bf16(
            false, a[am].v, false, b[bn].v, (short)0, acc[am][bn],
            false, false);
    __syncthreads();
  }

  // ---- store: C/D layout: VGPR v, lanes0-15 -> M=v, lanes16-31 -> M=v+8 ----
#pragma unroll
  for (int am = 0; am < 2; ++am) {
#pragma unroll
    for (int bn = 0; bn < 4; ++bn) {
      int col = colBase + wn * 64 + bn * 16 + lr;
#pragma unroll
      for (int v = 0; v < 8; ++v) {
        int row = blockRow + wm * 32 + am * 16 + lh * 8 + v;
        if (row < M) {
          float val = acc[am][bn][v];
          if (bias) val += bias[col];
          if (relu) val = fmaxf(val, 0.0f);
          size_t o = (size_t)row * Nn + col;
          if (Cf)  Cf[o]  = val;
          if (Cbf) Cbf[o] = f2bf(val);
        }
      }
    }
  }
}

// --------------------------------- launcher --------------------------------

extern "C" void kernel_launch(void* const* d_in, const int* in_sizes, int n_in,
                              void* d_out, int out_size, void* d_ws,
                              size_t ws_size, hipStream_t stream) {
  (void)in_sizes; (void)n_in; (void)out_size; (void)ws_size;

  // inputs (setup_inputs order)
  const float* x          = (const float*)d_in[0];
  const int*   edge_index = (const int*)d_in[1];
  const int*   edge_type  = (const int*)d_in[2];
  const int*   batch      = (const int*)d_in[3];
  const float* W_in       = (const float*)d_in[4];
  const float* b_in       = (const float*)d_in[5];
  const float* W_rel[3]   = {(const float*)d_in[6],  (const float*)d_in[11], (const float*)d_in[16]};
  const float* W_root[3]  = {(const float*)d_in[7],  (const float*)d_in[12], (const float*)d_in[17]};
  const float* b_conv[3]  = {(const float*)d_in[8],  (const float*)d_in[13], (const float*)d_in[18]};
  const float* gamma_[3]  = {(const float*)d_in[9],  (const float*)d_in[14], (const float*)d_in[19]};
  const float* beta_[3]   = {(const float*)d_in[10], (const float*)d_in[15], (const float*)d_in[20]};
  const float* W_c1       = (const float*)d_in[21];
  const float* b_c1       = (const float*)d_in[22];
  const float* W_c2       = (const float*)d_in[23];
  const float* b_c2       = (const float*)d_in[24];
  float* out = (float*)d_out;

  const int* e_src = edge_index;
  const int* e_dst = edge_index + EE;

  // workspace carve-up (256 B aligned)
  char* ws = (char*)d_ws;
  size_t off = 0;
  auto carve = [&](size_t bytes) -> char* {
    char* p = ws + off;
    off += (bytes + 255) & ~(size_t)255;
    return p;
  };
  const long ND  = (long)NN * DDC;
  float*          conv    = (float*)carve(ND * 4);
  float*          xinit   = (float*)carve(ND * 4);          // reused as h3
  float*          h1      = (float*)carve(ND * 4);
  float*          h2      = (float*)carve(ND * 4);
  unsigned short* hbuf    = (unsigned short*)carve(ND * 2); // current h, bf16
  unsigned short* Hr      = (unsigned short*)carve(ND * 2); // per-rel transform
  unsigned short* xbf     = (unsigned short*)carve((long)NN * DINC * 2);
  float*          cnt     = (float*)carve((long)RR * NN * 4);
  unsigned short* w_in_bf = (unsigned short*)carve((long)DINC * DDC * 2);
  unsigned short* w_rel_bf= (unsigned short*)carve((long)3 * RR * DDC * DDC * 2);
  unsigned short* w_rt_bf = (unsigned short*)carve((long)3 * DDC * DDC * 2);
  unsigned short* w_c1_bf = (unsigned short*)carve((long)DDC * 256 * 2);
  float*          bnsum   = (float*)carve(DDC * 4);
  float*          bnsq    = (float*)carve(DDC * 4);
  float*          psum    = (float*)carve((long)GGC * DDC * 4);
  float*          pcnt    = (float*)carve(GGC * 4);
  unsigned short* pooledbf= (unsigned short*)carve((long)GGC * DDC * 2);
  float*          hidden  = (float*)carve((long)GGC * 256 * 4);

  auto gblk = [](long n) { return (unsigned)((n + 255) / 256); };
  dim3 T(256);

  // ---- weight / input bf16 conversion ----
  k_cvt_bf16<<<gblk((long)NN * DINC), T, 0, stream>>>(x, xbf, (long)NN * DINC);
  k_cvt_bf16<<<gblk((long)DINC * DDC), T, 0, stream>>>(W_in, w_in_bf, (long)DINC * DDC);
  for (int l = 0; l < 3; ++l) {
    k_cvt_bf16<<<gblk((long)RR * DDC * DDC), T, 0, stream>>>(
        W_rel[l], w_rel_bf + (long)l * RR * DDC * DDC, (long)RR * DDC * DDC);
    k_cvt_bf16<<<gblk((long)DDC * DDC), T, 0, stream>>>(
        W_root[l], w_rt_bf + (long)l * DDC * DDC, (long)DDC * DDC);
  }
  k_cvt_bf16<<<gblk((long)DDC * 256), T, 0, stream>>>(W_c1, w_c1_bf, (long)DDC * 256);

  // ---- per-(relation,dst) mean normalizers (shared by all 3 layers) ----
  k_zero_f32<<<gblk((long)RR * NN), T, 0, stream>>>(cnt, (long)RR * NN);
  k_count_edges<<<gblk(EE), T, 0, stream>>>(e_dst, edge_type, cnt);
  k_inv_cnt<<<gblk((long)RR * NN), T, 0, stream>>>(cnt, (long)RR * NN);

  dim3 gemmBlk(256);
  auto gemmGrid = [](int M, int Nn) { return dim3((M + 127) / 128, Nn / 128); };

  // ---- x_init = x @ W_in + b_in  -> f32 (residual) + bf16 (layer input) ----
  k_gemm_bf16<<<gemmGrid(NN, DDC), gemmBlk, 0, stream>>>(
      xbf, w_in_bf, b_in, xinit, hbuf, NN, DDC, DINC, 0);

  // ---- three RGCN layers ----
  float* lay_out[3] = {h1, h2, xinit};   // xinit reused as h3 (safe: residual
  const float* res[3] = {nullptr, xinit, h1};  // reads complete before reuse)
  for (int l = 0; l < 3; ++l) {
    // root term: conv = h @ W_root + b   (plain store, scatter adds on top)
    k_gemm_bf16<<<gemmGrid(NN, DDC), gemmBlk, 0, stream>>>(
        hbuf, w_rt_bf + (long)l * DDC * DDC, b_conv[l], conv, nullptr,
        NN, DDC, DDC, 0);
    // relation terms: Hr = h @ W_r (bf16), scatter Hr[src]/cnt -> conv[dst]
    for (int r = 0; r < RR; ++r) {
      k_gemm_bf16<<<gemmGrid(NN, DDC), gemmBlk, 0, stream>>>(
          hbuf, w_rel_bf + ((long)l * RR + r) * DDC * DDC, nullptr,
          nullptr, Hr, NN, DDC, DDC, 0);
      k_scatter_rel<<<dim3(EE), T, 0, stream>>>(
          Hr, e_src, e_dst, edge_type, cnt, conv, r);
    }
    if (res[l])
      k_add_res<<<gblk(ND), T, 0, stream>>>(conv, res[l], ND);
    // BN (training-mode batch stats) + ReLU, emit f32 + bf16
    k_zero_f32<<<gblk(DDC), T, 0, stream>>>(bnsum, DDC);
    k_zero_f32<<<gblk(DDC), T, 0, stream>>>(bnsq, DDC);
    k_bn_stats<<<dim3(512), T, 0, stream>>>(conv, bnsum, bnsq);
    k_bn_apply_relu<<<gblk(ND), T, 0, stream>>>(
        conv, bnsum, bnsq, gamma_[l], beta_[l], lay_out[l], hbuf);
  }

  // ---- global mean pool ----
  k_zero_f32<<<gblk((long)GGC * DDC), T, 0, stream>>>(psum, (long)GGC * DDC);
  k_zero_f32<<<gblk(GGC), T, 0, stream>>>(pcnt, GGC);
  k_pool_accum<<<gblk(ND), T, 0, stream>>>(lay_out[2], batch, psum);
  k_pool_cnt<<<gblk(NN), T, 0, stream>>>(batch, pcnt);
  k_pool_div<<<gblk((long)GGC * DDC), T, 0, stream>>>(psum, pcnt, pooledbf);

  // ---- classifier: relu(pooled @ W_c1 + b_c1) @ W_c2 + b_c2 ----
  k_gemm_bf16<<<gemmGrid(GGC, 256), gemmBlk, 0, stream>>>(
      pooledbf, w_c1_bf, b_c1, hidden, nullptr, GGC, 256, DDC, 1);
  k_final_fc<<<dim3(1), dim3(128), 0, stream>>>(hidden, W_c2, b_c2, out);
}
